// TSA_300647710929
// MI455X (gfx1250) — matrix operand order repaired and checked by reading, not compiled, not run on hardware
//
#include <hip/hip_runtime.h>
#include <cstdint>
#include <cstddef>

// ---------------- types / constants ----------------
typedef __attribute__((ext_vector_type(16))) __bf16 v16bf;
typedef __attribute__((ext_vector_type(8)))  __bf16 v8bf;
typedef __attribute__((ext_vector_type(8)))  float  v8f;
typedef int v4i_ty __attribute__((vector_size(16)));

#define NI   32
#define CC   256
#define HWD  1024
#define NCHW (NI*CC*HWD)

#define KLD 40    // padded LDS row (bf16) for K tile rows of 32 cols: 80B, 16B-aligned
#define VLD 264   // padded LDS row (bf16) for V tile rows of 256 cols: 528B, 16B-aligned

__device__ __forceinline__ unsigned short f2bf(float x) {
    unsigned int u = __float_as_uint(x);
    unsigned int r = u + 0x7fffu + ((u >> 16) & 1u);   // round-to-nearest-even
    return (unsigned short)(r >> 16);
}

// A-operand (16x32 bf16, row-major src, stride ld):
// lanes 0-15: row=lane,   K halves {0..7, 16..23}
// lanes16-31: row=lane-16,K halves {8..15,24..31}
__device__ __forceinline__ v16bf load_A_frag(const __bf16* base, int ld, int lane) {
    int row = lane & 15;
    int kh  = (lane >> 4) << 3;               // 0 or 8
    const __bf16* p0 = base + (size_t)row * ld + kh;
    const __bf16* p1 = base + (size_t)row * ld + 16 + kh;
    v8bf lo = *(const v8bf*)p0;
    v8bf hi = *(const v8bf*)p1;
    return __builtin_shufflevector(lo, hi, 0,1,2,3,4,5,6,7,8,9,10,11,12,13,14,15);
}

// B-operand (32x16 bf16, row-major src, stride ld): lane l holds col N=l%16,
// K halves {0..7,16..23} (lanes 0-15) / {8..15,24..31} (lanes 16-31)
__device__ __forceinline__ v16bf load_B_frag(const __bf16* base, int ld, int lane) {
    int col = lane & 15;
    int kh  = (lane >> 4) << 3;
    v16bf r;
#pragma unroll
    for (int i = 0; i < 8; ++i) {
        r[i]     = base[(size_t)(kh + i) * ld + col];
        r[8 + i] = base[(size_t)(16 + kh + i) * ld + col];
    }
    return r;
}

#define WMMA_BF16(a,b,c) __builtin_amdgcn_wmma_f32_16x16x32_bf16(false,(a),false,(b),(short)0,(c),false,false)

// ---------------- CDNA5 async global->LDS (ASYNCcnt path) ----------------
__device__ __forceinline__ void async_g2l_b128(const void* g, void* l) {
#if __has_builtin(__builtin_amdgcn_global_load_async_to_lds_b128)
    __builtin_amdgcn_global_load_async_to_lds_b128(
        (__attribute__((address_space(1))) v4i_ty*)g,
        (__attribute__((address_space(3))) v4i_ty*)l, 0, 0);
#else
    unsigned loff = (unsigned)(size_t)(__attribute__((address_space(3))) void*)l;
    unsigned long long ga = (unsigned long long)(size_t)g;
    asm volatile("global_load_async_to_lds_b128 %0, %1, off"
                 :: "v"(loff), "v"(ga) : "memory");
#endif
}

__device__ __forceinline__ void wait_async0() {
#if __has_builtin(__builtin_amdgcn_s_wait_asynccnt)
    __builtin_amdgcn_s_wait_asynccnt(0);
#else
    asm volatile("s_wait_asynccnt 0x0" ::: "memory");
#endif
}

// ---------------- 0: cast x / weights to bf16 ----------------
__global__ void cast_kernel(const float* __restrict__ x,
                            const float* __restrict__ w1,
                            const float* __restrict__ w2,
                            const float* __restrict__ w0,
                            unsigned short* __restrict__ xb,
                            unsigned short* __restrict__ wb) {
    const int total = NCHW + 3 * CC * CC;
    for (int i = blockIdx.x * blockDim.x + threadIdx.x; i < total;
         i += gridDim.x * blockDim.x) {
        if (i < NCHW) {
            xb[i] = f2bf(x[i]);
        } else {
            int j = i - NCHW;
            float v = (j < CC*CC) ? w1[j] : (j < 2*CC*CC) ? w2[j - CC*CC] : w0[j - 2*CC*CC];
            wb[j] = f2bf(v);
        }
    }
}

// ---------------- 1: 1x1 convs via WMMA (with temporal shift) ----------------
// grid (NI*3, 2, 16), 256 threads (8 waves). Each wave: 16(o) x 64(p) tile.
__global__ __launch_bounds__(256) void conv_kernel(
    const __bf16* __restrict__ xb, const __bf16* __restrict__ wb,
    const float* __restrict__ b1, const float* __restrict__ b2, const float* __restrict__ b0,
    unsigned short* __restrict__ Qtmp, unsigned short* __restrict__ KB,
    unsigned short* __restrict__ Vtmp)
{
    const int lane = threadIdx.x & 31;
    const int wid  = threadIdx.x >> 5;
    const int which = blockIdx.x % 3;
    const int n     = blockIdx.x / 3;
    const int v = n >> 3, t = n & 7;

    int src; const float* bias; unsigned short* dst;
    if (which == 0)      { int ts = t - 1; src = v*8 + (ts > 0 ? ts : 0); bias = b1; dst = Qtmp; }
    else if (which == 1) { int ts = t - 2; src = v*8 + (ts > 0 ? ts : 0); bias = b2; dst = KB;   }
    else                 { src = n;                                       bias = b0; dst = Vtmp; }

    const __bf16* W = wb + (size_t)which * CC * CC;
    const __bf16* X = xb + (size_t)src * CC * HWD;
    dst += (size_t)n * CC * HWD;

    const int o0 = blockIdx.y * 128 + wid * 16;
    const int p0 = blockIdx.z * 64;

    v8f acc[4];
#pragma unroll
    for (int j = 0; j < 4; ++j) acc[j] = (v8f){0.f,0.f,0.f,0.f,0.f,0.f,0.f,0.f};

#pragma unroll
    for (int k0 = 0; k0 < CC; k0 += 32) {
        v16bf a = load_A_frag(W + (size_t)o0 * CC + k0, CC, lane);
#pragma unroll
        for (int j = 0; j < 4; ++j) {
            v16bf b = load_B_frag(X + (size_t)k0 * HWD + p0 + j*16, HWD, lane);
            acc[j] = WMMA_BF16(a, b, acc[j]);
        }
    }
#pragma unroll
    for (int j = 0; j < 4; ++j) {
#pragma unroll
        for (int r = 0; r < 8; ++r) {
            int o = o0 + r + ((lane >> 4) << 3);
            float val = acc[j][r] + bias[o];
            dst[(size_t)o * HWD + p0 + j*16 + (lane & 15)] = f2bf(val);
        }
    }
}

// ---------------- 2: bf16 [C][HW] -> [HW][C] transpose ----------------
__global__ __launch_bounds__(256) void transpose_kernel(const unsigned short* __restrict__ src,
                                                        unsigned short* __restrict__ dst) {
    __shared__ unsigned short tile[32][33];
    const int n  = blockIdx.x;
    const int c0 = blockIdx.y * 32, p0 = blockIdx.z * 32;
    const unsigned short* s = src + (size_t)n * CC * HWD;
    unsigned short* d = dst + (size_t)n * HWD * CC;
    const int tx = threadIdx.x & 31, ty = threadIdx.x >> 5;
#pragma unroll
    for (int i = 0; i < 4; ++i) { int r = ty + i*8; tile[r][tx] = s[(size_t)(c0 + r) * HWD + p0 + tx]; }
    __syncthreads();
#pragma unroll
    for (int i = 0; i < 4; ++i) { int r = ty + i*8; d[(size_t)(p0 + r) * CC + c0 + tx] = tile[tx][r]; }
}

// ---------------- 3: flash attention with async-LDS staged K/V tiles ----------------
// grid (NI, 8), 256 threads (8 waves). One wave per 16 query rows.
// Per g-chunk (32 cols): K tile [256c x 32g] and V tile [32g x 256c] staged once
// per block via GLOBAL_LOAD_ASYNC_TO_LDS, shared by all 8 waves.
__global__ __launch_bounds__(256) void attn_kernel(
    const __bf16* __restrict__ QT, const __bf16* __restrict__ KB,
    const __bf16* __restrict__ VT, float* __restrict__ OUTB)
{
    __shared__ unsigned short ldsK[256 * KLD];     // 20480 B
    __shared__ unsigned short ldsV[32 * VLD];      // 16896 B
    __shared__ unsigned short ldsP[8 * 16 * 32];   // per-wave P staging, 8192 B
    const int tid  = threadIdx.x;
    const int lane = tid & 31;
    const int wid  = tid >> 5;
    const int n    = blockIdx.x;
    const int t0   = (blockIdx.y * 8 + wid) * 16;

    const __bf16* Qn = QT + (size_t)n * HWD * CC;   // [t][c]
    const __bf16* Kn = KB + (size_t)n * CC * HWD;   // [c][g]
    const __bf16* Vn = VT + (size_t)n * HWD * CC;   // [g][c]
    float* On = OUTB + (size_t)n * HWD * CC;        // [t][c]

    const __bf16* ldsKb = (const __bf16*)ldsK;
    const __bf16* ldsVb = (const __bf16*)ldsV;

    v16bf qa[8];
#pragma unroll
    for (int kk = 0; kk < 8; ++kk)
        qa[kk] = load_A_frag(Qn + (size_t)t0 * CC + kk * 32, CC, lane);

    v8f acc[16];
#pragma unroll
    for (int i = 0; i < 16; ++i) acc[i] = (v8f){0.f,0.f,0.f,0.f,0.f,0.f,0.f,0.f};
    float m[8], lsum[8];
#pragma unroll
    for (int r = 0; r < 8; ++r) { m[r] = -__builtin_inff(); lsum[r] = 0.f; }

    unsigned short* pl = ldsP + wid * (16 * 32);
    const __bf16* plb = (const __bf16*)pl;
    const float scale = 0.0625f;                    // C^-0.5 = 1/16

    for (int g0 = 0; g0 < HWD; g0 += 32) {
        // ---- cooperative async stage of K/V tiles (1024 x 16B chunks each) ----
#pragma unroll
        for (int i = 0; i < 4; ++i) {               // K: row c, 4 chunks of 8 bf16
            int chunk = i * 256 + tid;
            int c = chunk >> 2, h = chunk & 3;
            async_g2l_b128(Kn + (size_t)c * HWD + g0 + h * 8, ldsK + c * KLD + h * 8);
        }
#pragma unroll
        for (int i = 0; i < 4; ++i) {               // V: row g, 32 chunks of 8 bf16
            int chunk = i * 256 + tid;
            int g = chunk >> 5, h = chunk & 31;
            async_g2l_b128(Vn + (size_t)(g0 + g) * CC + h * 8, ldsV + g * VLD + h * 8);
        }
        wait_async0();
        __syncthreads();                            // tiles ready for all waves

        // ---- S = (Q^T K) tile, 16 x 32, from LDS ----
        v8f s0 = (v8f){0.f,0.f,0.f,0.f,0.f,0.f,0.f,0.f};
        v8f s1 = (v8f){0.f,0.f,0.f,0.f,0.f,0.f,0.f,0.f};
#pragma unroll
        for (int kk = 0; kk < 8; ++kk) {
            v16bf bk0 = load_B_frag(ldsKb + (size_t)(kk*32) * KLD,      KLD, lane);
            v16bf bk1 = load_B_frag(ldsKb + (size_t)(kk*32) * KLD + 16, KLD, lane);
            s0 = WMMA_BF16(qa[kk], bk0, s0);
            s1 = WMMA_BF16(qa[kk], bk1, s1);
        }
        // ---- online softmax (rows live per (VGPR r, lane-half)) ----
        float f[8];
#pragma unroll
        for (int r = 0; r < 8; ++r) {
            float a0 = s0[r] * scale, a1 = s1[r] * scale;
            float mx = fmaxf(a0, a1);
            mx = fmaxf(mx, __shfl_xor(mx, 1, 32));
            mx = fmaxf(mx, __shfl_xor(mx, 2, 32));
            mx = fmaxf(mx, __shfl_xor(mx, 4, 32));
            mx = fmaxf(mx, __shfl_xor(mx, 8, 32));
            float mnew = fmaxf(m[r], mx);
            f[r] = __expf(m[r] - mnew);
            m[r] = mnew;
            float p0v = __expf(a0 - mnew);
            float p1v = __expf(a1 - mnew);
            s0[r] = p0v; s1[r] = p1v;
            float rs = p0v + p1v;
            rs += __shfl_xor(rs, 1, 32);
            rs += __shfl_xor(rs, 2, 32);
            rs += __shfl_xor(rs, 4, 32);
            rs += __shfl_xor(rs, 8, 32);
            lsum[r] = lsum[r] * f[r] + rs;
        }
        // ---- stage P (D-layout -> row-major [16][32] bf16) in per-wave LDS ----
#pragma unroll
        for (int r = 0; r < 8; ++r) {
            int trow = r + ((lane >> 4) << 3);
            pl[trow*32 +      (lane & 15)] = f2bf(s0[r]);
            pl[trow*32 + 16 + (lane & 15)] = f2bf(s1[r]);
        }
        asm volatile("s_wait_dscnt 0x0" ::: "memory");
        v16bf pa = load_A_frag(plb, 32, lane);
        // ---- out += P * V^T, V from LDS ----
#pragma unroll
        for (int ct = 0; ct < 16; ++ct) {
#pragma unroll
            for (int r = 0; r < 8; ++r) acc[ct][r] *= f[r];
            v16bf vb = load_B_frag(ldsVb + ct * 16, VLD, lane);
            acc[ct] = WMMA_BF16(pa, vb, acc[ct]);
        }
        __syncthreads();                            // readers drained before next stage
    }
    float inv[8];
#pragma unroll
    for (int r = 0; r < 8; ++r) inv[r] = 1.0f / lsum[r];
#pragma unroll
    for (int ct = 0; ct < 16; ++ct) {
#pragma unroll
        for (int r = 0; r < 8; ++r) {
            int t = t0 + r + ((lane >> 4) << 3);
            On[(size_t)t * CC + ct*16 + (lane & 15)] = acc[ct][r] * inv[r];
        }
    }
}

// ---------------- 4: BN statistics (two-stage, deterministic) ----------------
__global__ __launch_bounds__(256) void bn_partial_kernel(const float* __restrict__ outb,
                                                         float* __restrict__ part) {
    const int c = threadIdx.x;
    const int base = blockIdx.x * 256;          // 128 blocks x 256 rows = 32768 rows
    float s = 0.f, q = 0.f;
    for (int r = 0; r < 256; ++r) {
        float v = outb[(size_t)(base + r) * CC + c];
        s += v; q += v * v;
    }
    part[(size_t)blockIdx.x * 512 + c]       = s;
    part[(size_t)blockIdx.x * 512 + 256 + c] = q;
}

__global__ __launch_bounds__(256) void bn_reduce_kernel(const float* __restrict__ part,
                                                        float* __restrict__ mv) {
    const int c = threadIdx.x;
    float s = 0.f, q = 0.f;
    for (int i = 0; i < 128; ++i) {
        s += part[(size_t)i * 512 + c];
        q += part[(size_t)i * 512 + 256 + c];
    }
    const float inv_n = 1.0f / 32768.0f;
    float mean = s * inv_n;
    float var  = q * inv_n - mean * mean;
    mv[c]       = mean;
    mv[256 + c] = rsqrtf(var + 1e-5f);
}

// ---------------- 5: BN affine + residual, transpose to [N][C][HW] ----------------
__global__ __launch_bounds__(256) void final_kernel(const float* __restrict__ x,
                                                    const float* __restrict__ outb,
                                                    const float* __restrict__ mv,
                                                    const float* __restrict__ gamma,
                                                    const float* __restrict__ beta,
                                                    float* __restrict__ y) {
    const int n = blockIdx.x, t0 = blockIdx.y * 32, c = threadIdx.x;
    const float mu = mv[c], is = mv[256 + c];
    const float a = is * gamma[c];
    const float b = beta[c] - mu * a;
    float vals[32];
#pragma unroll
    for (int i = 0; i < 32; ++i)        // coalesced reads from [N][HW][C]
        vals[i] = outb[((size_t)n * HWD + t0 + i) * CC + c];
#pragma unroll
    for (int i = 0; i < 32; ++i) {      // each thread: 128B contiguous store
        size_t xi = ((size_t)n * CC + c) * HWD + t0 + i;
        y[xi] = x[xi] + vals[i] * a + b;
    }
}

// ---------------- host launcher ----------------
extern "C" void kernel_launch(void* const* d_in, const int* in_sizes, int n_in,
                              void* d_out, int out_size, void* d_ws, size_t ws_size,
                              hipStream_t stream) {
    (void)in_sizes; (void)n_in; (void)out_size; (void)ws_size;
    const float* x  = (const float*)d_in[0];
    const float* w1 = (const float*)d_in[1];
    const float* b1 = (const float*)d_in[2];
    const float* w2 = (const float*)d_in[3];
    const float* b2 = (const float*)d_in[4];
    const float* w0 = (const float*)d_in[5];
    const float* b0 = (const float*)d_in[6];
    const float* gamma = (const float*)d_in[7];
    const float* beta  = (const float*)d_in[8];
    float* y = (float*)d_out;

    char* ws = (char*)d_ws;
    size_t p = 0;
    auto take = [&](size_t bytes) -> char* {
        p = (p + 255) & ~(size_t)255;
        char* r = ws + p;
        p += bytes;
        return r;
    };
    unsigned short* XB   = (unsigned short*)take((size_t)NCHW * 2);     // xb; reused as QT
    unsigned short* WB   = (unsigned short*)take((size_t)3 * CC * CC * 2);
    unsigned short* KB   = (unsigned short*)take((size_t)NCHW * 2);
    unsigned short* QTMP = (unsigned short*)take((size_t)NCHW * 2);     // reused as VT
    unsigned short* VTMP = (unsigned short*)take((size_t)NCHW * 2);
    float* OUTB = (float*)take((size_t)NCHW * 4);
    float* PART = (float*)take((size_t)128 * 512 * 4);
    float* MV   = (float*)take((size_t)512 * 4);
    unsigned short* QT = XB;     // xb dead after conv_kernel (stream-ordered)
    unsigned short* VT = QTMP;   // Qtmp dead after first transpose (stream-ordered)

    cast_kernel<<<4096, 256, 0, stream>>>(x, w1, w2, w0, XB, WB);
    conv_kernel<<<dim3(NI * 3, 2, 16), 256, 0, stream>>>(
        (const __bf16*)XB, (const __bf16*)WB, b1, b2, b0, QTMP, KB, VTMP);
    transpose_kernel<<<dim3(NI, 8, 32), 256, 0, stream>>>(QTMP, QT);
    transpose_kernel<<<dim3(NI, 8, 32), 256, 0, stream>>>(VTMP, VT);
    attn_kernel<<<dim3(NI, 8), 256, 0, stream>>>(
        (const __bf16*)QT, (const __bf16*)KB, (const __bf16*)VT, OUTB);
    bn_partial_kernel<<<128, 256, 0, stream>>>(OUTB, PART);
    bn_reduce_kernel<<<1, 256, 0, stream>>>(PART, MV);
    final_kernel<<<dim3(NI, 32), 256, 0, stream>>>(x, OUTB, MV, gamma, beta, y);
}
